// SimpleRNN_29901562314752
// MI455X (gfx1250) — compile-verified, matrix operand order checked by reading, not run on hardware
//
#include <hip/hip_runtime.h>

#define DT_TAU 0.05f
#define TSQ    256
#define BATCH  64
#define NIN    512
#define HID    2048
#define NOUT   512

typedef __attribute__((ext_vector_type(16))) __bf16 v16bf;
typedef __attribute__((ext_vector_type(8)))  __bf16 v8bf;
typedef __attribute__((ext_vector_type(8)))  float  v8f;

static __device__ __forceinline__ __bf16 f32_to_bf16(float f) {
    unsigned u = __builtin_bit_cast(unsigned, f);
    unsigned r = u + 0x7FFFu + ((u >> 16) & 1u);   // round-to-nearest-even
    unsigned short hbits = (unsigned short)(r >> 16);
    return __builtin_bit_cast(__bf16, hbits);
}

// A-operand loader: row-major bf16 [M x lda] -> wave32 WMMA 16x32 A layout.
// Lane l (0-15): row m0+l, K chunks [k0+0..7] and [k0+16..23];
// Lane l+16   : row m0+l, K chunks [k0+8..15] and [k0+24..31].
static __device__ __forceinline__ v16bf load_a_bf16(const __bf16* __restrict__ A, int lda,
                                                    int m0, int k0, int lane) {
    int m  = m0 + (lane & 15);
    int kb = k0 + ((lane >> 4) << 3);
    const __bf16* p = A + (size_t)m * lda + kb;
    v8bf lo = *(const v8bf*)p;            // global_load_b128
    v8bf hi = *(const v8bf*)(p + 16);     // global_load_b128
    return __builtin_shufflevector(lo, hi, 0,1,2,3,4,5,6,7,8,9,10,11,12,13,14,15);
}

static __device__ __forceinline__ v8f wmma_bf16(v16bf a, v16bf b, v8f c) {
    return __builtin_amdgcn_wmma_f32_16x16x32_bf16(false, a, false, b, (short)0, c, false, false);
}

// ---------------- prep kernels ----------------

__global__ void k_cvt_bf16(const float* __restrict__ src, __bf16* __restrict__ dst, int n) {
    int i = (blockIdx.x * blockDim.x + threadIdx.x) * 4;
    if (i + 3 < n) {
        float4 v = *(const float4*)(src + i);
        dst[i + 0] = f32_to_bf16(v.x);
        dst[i + 1] = f32_to_bf16(v.y);
        dst[i + 2] = f32_to_bf16(v.z);
        dst[i + 3] = f32_to_bf16(v.w);
    }
}

// Pack row-major f32 B [K x N] into WMMA B-register layout:
// dst[((nt*kT + kt)*32 + lane)] = 16 bf16 values this lane feeds to v_wmma.
__global__ void k_pack_b(const float* __restrict__ src, v16bf* __restrict__ dst, int K, int N) {
    int lane = threadIdx.x & 31;
    int wave = blockIdx.x * (blockDim.x >> 5) + (threadIdx.x >> 5);
    int kT = K >> 5;
    int nt = wave / kT;
    int kt = wave - nt * kT;
    int n  = (nt << 4) + (lane & 15);
    int kb = (kt << 5) + ((lane >> 4) << 3);
    v16bf v;
#pragma unroll
    for (int i = 0; i < 8; ++i) {
        v[i]     = f32_to_bf16(src[(size_t)(kb + i) * N + n]);
        v[8 + i] = f32_to_bf16(src[(size_t)(kb + 16 + i) * N + n]);
    }
    dst[(size_t)wave * 32 + lane] = v;
}

__global__ void k_init(float* __restrict__ h, __bf16* __restrict__ a, int n) {
    int i = blockIdx.x * blockDim.x + threadIdx.x;
    if (i < n) {
        h[i] = 0.0f;
        a[i] = __builtin_bit_cast(__bf16, (unsigned short)0);
    }
}

// ---------------- input projection: tot_input[b][t][h] = (x @ Win)[t*64+b][h] ----------------

__global__ void k_input_proj(const __bf16* __restrict__ xb,   // [TSQ*BATCH][NIN] row-major bf16
                             const v16bf* __restrict__ Wpk,   // packed Win
                             float* __restrict__ totin) {     // [BATCH][TSQ][HID]
    int lane = threadIdx.x & 31;
    int wave = blockIdx.x * (blockDim.x >> 5) + (threadIdx.x >> 5);
    int nt = wave & 127;               // HID/16 = 128 tiles
    int mt = wave >> 7;                // (TSQ*BATCH)/16 = 1024 tiles
    int m0 = mt << 4;
    const int kT = NIN / 32;           // 16
    v8f c = {};
    for (int kt = 0; kt < kT; ++kt) {
        v16bf a = load_a_bf16(xb, NIN, m0, kt * 32, lane);
        v16bf b = Wpk[((size_t)nt * kT + kt) * 32 + lane];
        c = wmma_bf16(a, b, c);
    }
    int col   = (nt << 4) + (lane & 15);
    int rbase = m0 + ((lane >> 4) << 3);
#pragma unroll
    for (int r = 0; r < 8; ++r) {
        int row = rbase + r;           // global row = t*BATCH + b
        int t = row >> 6, b = row & 63;
        totin[((size_t)b * TSQ + t) * HID + col] = c[r];
    }
}

// ---------------- recurrent step: h' = (1-dt)*h + dt*(tanh(h)@Wr + inp + bias) ----------------

__global__ void k_rnn_step(const __bf16* __restrict__ Ain,   // tanh(h_t)  [BATCH][HID] bf16
                           const v16bf* __restrict__ WrPk,   // packed Wr
                           const float* __restrict__ totin,  // [BATCH][TSQ][HID]
                           const float* __restrict__ bias,   // [HID]
                           float* __restrict__ h,            // [BATCH][HID] f32 (in/out)
                           float* __restrict__ rnnhid,       // [BATCH][TSQ][HID]
                           __bf16* __restrict__ Aout,        // tanh(h_{t+1}) bf16
                           int t) {
    int lane = threadIdx.x & 31;
    int wave = blockIdx.x * (blockDim.x >> 5) + (threadIdx.x >> 5);  // 0..511
    int nt = wave >> 2;                // 0..127
    int m0 = (wave & 3) << 4;
    const int kT = HID / 32;           // 64
    v8f c = {};
    for (int kt = 0; kt < kT; ++kt) {
        v16bf a = load_a_bf16(Ain, HID, m0, kt * 32, lane);
        v16bf b = WrPk[((size_t)nt * kT + kt) * 32 + lane];
        c = wmma_bf16(a, b, c);
    }
    int col   = (nt << 4) + (lane & 15);
    int rbase = m0 + ((lane >> 4) << 3);
    float bv  = bias[col];
#pragma unroll
    for (int r = 0; r < 8; ++r) {
        int row   = rbase + r;         // batch index
        size_t hi = (size_t)row * HID + col;
        float inp = totin[((size_t)row * TSQ + t) * HID + col];
        float hn  = (1.0f - DT_TAU) * h[hi] + DT_TAU * (c[r] + inp + bv);
        h[hi] = hn;
        rnnhid[((size_t)row * TSQ + t) * HID + col] = hn;
        Aout[hi] = f32_to_bf16(tanhf(hn));
    }
}

// ---------------- output projection: out_t = tanh(h_{t+1}) @ Wout ----------------

__global__ void k_rnn_out(const __bf16* __restrict__ A,      // tanh(h_{t+1}) [BATCH][HID]
                          const v16bf* __restrict__ WoPk,    // packed Wout
                          float* __restrict__ outs,          // [TSQ][BATCH][NOUT]
                          float* __restrict__ totout,        // [BATCH][TSQ][NOUT]
                          int t) {
    int lane = threadIdx.x & 31;
    int wave = blockIdx.x * (blockDim.x >> 5) + (threadIdx.x >> 5);  // 0..127
    int nt = wave >> 2;                // 0..31
    int m0 = (wave & 3) << 4;
    const int kT = HID / 32;           // 64
    v8f c = {};
    for (int kt = 0; kt < kT; ++kt) {
        v16bf a = load_a_bf16(A, HID, m0, kt * 32, lane);
        v16bf b = WoPk[((size_t)nt * kT + kt) * 32 + lane];
        c = wmma_bf16(a, b, c);
    }
    int col   = (nt << 4) + (lane & 15);
    int rbase = m0 + ((lane >> 4) << 3);
#pragma unroll
    for (int r = 0; r < 8; ++r) {
        int row = rbase + r;
        float v = c[r];
        outs[((size_t)t * BATCH + row) * NOUT + col] = v;
        totout[((size_t)row * TSQ + t) * NOUT + col] = v;
    }
}

extern "C" void kernel_launch(void* const* d_in, const int* in_sizes, int n_in,
                              void* d_out, int out_size, void* d_ws, size_t ws_size,
                              hipStream_t stream) {
    const float* x    = (const float*)d_in[0];
    const float* Win  = (const float*)d_in[1];
    const float* Wr   = (const float*)d_in[2];
    const float* bias = (const float*)d_in[3];
    const float* Wout = (const float*)d_in[4];

    float* out    = (float*)d_out;
    float* outs   = out;                                    // [TSQ][BATCH][NOUT]
    float* totin  = outs   + (size_t)TSQ * BATCH * NOUT;    // [BATCH][TSQ][HID]
    float* rnnhid = totin  + (size_t)BATCH * TSQ * HID;     // [BATCH][TSQ][HID]
    float* totout = rnnhid + (size_t)BATCH * TSQ * HID;     // [BATCH][TSQ][NOUT]

    char* ws = (char*)d_ws;
    v16bf* WrPk  = (v16bf*)ws;  ws += (size_t)HID  * HID  * 2;   // 8 MB
    v16bf* WinPk = (v16bf*)ws;  ws += (size_t)NIN  * HID  * 2;   // 2 MB
    v16bf* WoPk  = (v16bf*)ws;  ws += (size_t)HID  * NOUT * 2;   // 2 MB
    __bf16* xb   = (__bf16*)ws; ws += (size_t)TSQ * BATCH * NIN * 2; // 16 MB
    float*  h    = (float*)ws;  ws += (size_t)BATCH * HID * 4;   // 512 KB
    __bf16* A0   = (__bf16*)ws; ws += (size_t)BATCH * HID * 2;   // 256 KB
    __bf16* A1   = (__bf16*)ws;                                   // 256 KB

    // one-time prep: bf16 conversion + B-operand packing
    k_cvt_bf16<<<(TSQ * BATCH * NIN) / 1024, 256, 0, stream>>>(x, xb, TSQ * BATCH * NIN);
    k_pack_b<<<((NIN / 32) * (HID / 16)) / 8, 256, 0, stream>>>(Win, WinPk, NIN, HID);
    k_pack_b<<<((HID / 32) * (HID / 16)) / 8, 256, 0, stream>>>(Wr, WrPk, HID, HID);
    k_pack_b<<<((HID / 32) * (NOUT / 16)) / 8, 256, 0, stream>>>(Wout, WoPk, HID, NOUT);
    k_init<<<(BATCH * HID) / 256, 256, 0, stream>>>(h, A0, BATCH * HID);

    // input projection (also materializes tot_input output)
    k_input_proj<<<((TSQ * BATCH / 16) * (HID / 16)) / 8, 256, 0, stream>>>(xb, WinPk, totin);

    // sequential recurrence: 256 steps, double-buffered tanh activations
    __bf16* bufs[2] = {A0, A1};
    for (int t = 0; t < TSQ; ++t) {
        __bf16* Ain  = bufs[t & 1];
        __bf16* Aout = bufs[(t + 1) & 1];
        k_rnn_step<<<64, 256, 0, stream>>>(Ain, WrPk, totin, bias, h, rnnhid, Aout, t);
        k_rnn_out<<<16, 256, 0, stream>>>(Aout, WoPk, outs, totout, t);
    }
}
